// SpikeGRU4Rec_26156350832992
// MI455X (gfx1250) — compile-verified, matrix operand order checked by reading, not run on hardware
//
#include <hip/hip_runtime.h>
#include <hip/hip_bf16.h>

// ---------------------------------------------------------------------------
// SpikeGRU4Rec on MI455X (gfx1250): f16 WMMA GEMMs for the spiking GRU gates.
// B=256, L=200, D=512, T=4. All gate inputs are exact in f16 (spikes are
// binary, snn values are multiples of 0.25), outputs are sign-thresholded,
// so v_wmma_f32_16x16x32_f16 with f32 accumulation is numerically safe.
// K-loops are split at the x|h boundary (no per-iter selects) and software
// pipelined (preload k+1 fragments before issuing k's WMMAs).
// ---------------------------------------------------------------------------

typedef _Float16 v16h __attribute__((ext_vector_type(16)));
typedef _Float16 v8h  __attribute__((ext_vector_type(8)));
typedef float    v8f  __attribute__((ext_vector_type(8)));

union V16 { v16h v; v8h h[2]; };

#define BB 256   // batch
#define LL 200   // sequence length
#define DD 512   // hidden dim
#define NT 32    // DD/16 N-tiles

#define WMMA_F16(A, B, C) \
    __builtin_amdgcn_wmma_f32_16x16x32_f16(false, (A), false, (B), (short)0, (C), false, false)

// ---------------------------------------------------------------------------
// Pack an f32 row-major weight W[N][K] (torch Linear: out x in) into
// fragment-ready f16 tiles P[K/32][N/16][32][16] matching the 16-bit WMMA
// B-operand lane layout: lane l holds column n = l%16, K pattern
// k0 + {0..7} and k0 + 16 + {0..7}, k0 = (l<16 ? 0 : 8).
// ---------------------------------------------------------------------------
__global__ void pack_w_kernel(const float* __restrict__ W,
                              _Float16* __restrict__ P, int K, int N) {
    int gid = blockIdx.x * blockDim.x + threadIdx.x;
    int total = K * N;
    if (gid >= total) return;
    int i    = gid & 15;
    int lane = (gid >> 4) & 31;
    int nt   = (gid >> 9) % (N >> 4);
    int kt   = gid / (512 * (N >> 4));
    int k0   = (lane < 16) ? 0 : 8;
    int kk   = kt * 32 + k0 + ((i < 8) ? i : (i + 8));
    int nn   = nt * 16 + (lane & 15);
    P[gid] = (_Float16)W[(size_t)nn * K + kk];
}

__global__ void zero_h16_kernel(_Float16* p, int n) {
    int i = blockIdx.x * blockDim.x + threadIdx.x;
    if (i < n) p[i] = (_Float16)0.0f;
}

// ---------------------------------------------------------------------------
// Embedding gather + LayerNorm + 4-step LIF (soft reset), write snn as f16
// in [l][b][d] layout so each GRU timestep reads a contiguous [B][D] slab.
// One block per (b,l); 256 threads x 2 elements.
// ---------------------------------------------------------------------------
__global__ __launch_bounds__(256)
void embed_ln_lif_kernel(const int* __restrict__ seq,
                         const float* __restrict__ emb,
                         const float* __restrict__ gamma,
                         const float* __restrict__ beta,
                         _Float16* __restrict__ snn) {
    int bid = blockIdx.x;           // 0 .. B*L-1
    int b = bid % BB;
    int l = bid / BB;
    int tid = threadIdx.x;
    int item = seq[(size_t)b * LL + l];
    const float* row = emb + (size_t)item * DD;
    float x0 = row[tid];
    float x1 = row[tid + 256];

    __shared__ float ssum[256];
    __shared__ float ssq[256];
    ssum[tid] = x0 + x1;
    ssq[tid]  = x0 * x0 + x1 * x1;
    __syncthreads();
    for (int s = 128; s > 0; s >>= 1) {
        if (tid < s) { ssum[tid] += ssum[tid + s]; ssq[tid] += ssq[tid + s]; }
        __syncthreads();
    }
    float mu   = ssum[0] * (1.0f / DD);
    float var  = ssq[0] * (1.0f / DD) - mu * mu;
    float rstd = rsqrtf(var + 1e-5f);

    float xn0 = (x0 - mu) * rstd * gamma[tid]       + beta[tid];
    float xn1 = (x1 - mu) * rstd * gamma[tid + 256] + beta[tid + 256];

    // LIF: v += (x - v)/2 ; spike if v >= 1 ; soft reset v -= 1 ; mean over T=4
    float out0 = 0.0f, out1 = 0.0f, v0 = 0.0f, v1 = 0.0f;
#pragma unroll
    for (int t = 0; t < 4; ++t) {
        v0 += (xn0 - v0) * 0.5f;
        v1 += (xn1 - v1) * 0.5f;
        if (v0 >= 1.0f) { out0 += 0.25f; v0 -= 1.0f; }
        if (v1 >= 1.0f) { out1 += 0.25f; v1 -= 1.0f; }
    }
    size_t o = ((size_t)l * BB + b) * DD;
    snn[o + tid]       = (_Float16)out0;
    snn[o + tid + 256] = (_Float16)out1;
}

// ---------------------------------------------------------------------------
// GRU step part 1: r = (comb@Wr.T >= 0), z = (comb@Wz.T >= 0)
// comb = [x_t | h_prev], K = 2D = 1024 (2 halves x 16 k-tiles).
// grid (B/16, D/64), 4 waves/block, one 16x16 N-tile per wave, two gates.
// ---------------------------------------------------------------------------
__global__ __launch_bounds__(128)
void gru_rz_kernel(const _Float16* __restrict__ xall, int t,
                   const _Float16* __restrict__ hprev,
                   const _Float16* __restrict__ WrP,
                   const _Float16* __restrict__ WzP,
                   _Float16* __restrict__ rbuf,
                   _Float16* __restrict__ zbuf) {
    int lane = threadIdx.x & 31;
    int wave = threadIdx.x >> 5;
    int m0 = blockIdx.x * 16;
    int n0 = blockIdx.y * 64 + wave * 16;
    int nt = n0 >> 4;
    int mrowA = m0 + (lane & 15);
    int k0 = (lane < 16) ? 0 : 8;
    const _Float16* xrow = xall + ((size_t)t * BB + mrowA) * DD + k0;
    const _Float16* hrow = hprev + (size_t)mrowA * DD + k0;

    v8f accR = {};
    v8f accZ = {};

    // software-pipelined half: 16 k-tiles off a fixed base pointer
    auto run_half = [&](const _Float16* abase, int ktBase) {
        V16 a;
        a.h[0] = *(const v8h*)(abase);
        a.h[1] = *(const v8h*)(abase + 16);
        size_t bo = (((size_t)ktBase * NT + nt) * 32 + lane) * 16;
        v16h bR = *(const v16h*)(WrP + bo);
        v16h bZ = *(const v16h*)(WzP + bo);
#pragma unroll
        for (int kk = 0; kk < 16; ++kk) {
            V16 an;
            v16h bRn, bZn;
            if (kk < 15) {                       // preload k+1 fragments
                const _Float16* ar = abase + (kk + 1) * 32;
                an.h[0] = *(const v8h*)(ar);
                an.h[1] = *(const v8h*)(ar + 16);
                size_t bon = (((size_t)(ktBase + kk + 1) * NT + nt) * 32 + lane) * 16;
                bRn = *(const v16h*)(WrP + bon);
                bZn = *(const v16h*)(WzP + bon);
            }
            accR = WMMA_F16(a.v, bR, accR);
            accZ = WMMA_F16(a.v, bZ, accZ);
            if (kk < 15) { a = an; bR = bRn; bZ = bZn; }
        }
    };
    run_half(xrow, 0);    // K = 0..511   (x_t)
    run_half(hrow, 16);   // K = 512..1023 (h_prev)

    int mext = (lane >= 16) ? 8 : 0;
    int ncol = n0 + (lane & 15);
#pragma unroll
    for (int i = 0; i < 8; ++i) {
        size_t idx = (size_t)(m0 + i + mext) * DD + ncol;
        rbuf[idx] = (accR[i] >= 0.0f) ? (_Float16)1.0f : (_Float16)0.0f;
        zbuf[idx] = (accZ[i] >= 0.0f) ? (_Float16)1.0f : (_Float16)0.0f;
    }
}

// ---------------------------------------------------------------------------
// GRU step part 2: n = ([x_t | r*h]@Wn.T >= 0), h_new = z ? h : n (binary),
// plus gather h_new into hg where t == item_seq_len[b]-1.
// ---------------------------------------------------------------------------
__global__ __launch_bounds__(128)
void gru_nh_kernel(const _Float16* __restrict__ xall, int t,
                   const _Float16* __restrict__ hprev,
                   const _Float16* __restrict__ rbuf,
                   const _Float16* __restrict__ zbuf,
                   const _Float16* __restrict__ WnP,
                   _Float16* __restrict__ hnext,
                   _Float16* __restrict__ hg,
                   const int* __restrict__ lens) {
    int lane = threadIdx.x & 31;
    int wave = threadIdx.x >> 5;
    int m0 = blockIdx.x * 16;
    int n0 = blockIdx.y * 64 + wave * 16;
    int nt = n0 >> 4;
    int mrowA = m0 + (lane & 15);
    int k0 = (lane < 16) ? 0 : 8;
    const _Float16* xrow = xall + ((size_t)t * BB + mrowA) * DD + k0;
    const _Float16* hrow = hprev + (size_t)mrowA * DD + k0;
    const _Float16* rrow = rbuf + (size_t)mrowA * DD + k0;

    v8f accN = {};

    // half 1: x_t, pipelined
    {
        V16 a;
        a.h[0] = *(const v8h*)(xrow);
        a.h[1] = *(const v8h*)(xrow + 16);
        size_t bo = (((size_t)0 * NT + nt) * 32 + lane) * 16;
        v16h bN = *(const v16h*)(WnP + bo);
#pragma unroll
        for (int kk = 0; kk < 16; ++kk) {
            V16 an;
            v16h bNn;
            if (kk < 15) {
                const _Float16* ar = xrow + (kk + 1) * 32;
                an.h[0] = *(const v8h*)(ar);
                an.h[1] = *(const v8h*)(ar + 16);
                size_t bon = (((size_t)(kk + 1) * NT + nt) * 32 + lane) * 16;
                bNn = *(const v16h*)(WnP + bon);
            }
            accN = WMMA_F16(a.v, bN, accN);
            if (kk < 15) { a = an; bN = bNn; }
        }
    }
    // half 2: r*h (v_pk_mul_f16 on the fly), pipelined
    {
        V16 a;
        {
            v8h r0 = *(const v8h*)(rrow), r1 = *(const v8h*)(rrow + 16);
            v8h h0 = *(const v8h*)(hrow), h1 = *(const v8h*)(hrow + 16);
            a.h[0] = r0 * h0;
            a.h[1] = r1 * h1;
        }
        size_t bo = (((size_t)16 * NT + nt) * 32 + lane) * 16;
        v16h bN = *(const v16h*)(WnP + bo);
#pragma unroll
        for (int kk = 0; kk < 16; ++kk) {
            V16 an;
            v16h bNn;
            if (kk < 15) {
                const _Float16* rr = rrow + (kk + 1) * 32;
                const _Float16* hh = hrow + (kk + 1) * 32;
                v8h r0 = *(const v8h*)(rr), r1 = *(const v8h*)(rr + 16);
                v8h h0 = *(const v8h*)(hh), h1 = *(const v8h*)(hh + 16);
                an.h[0] = r0 * h0;
                an.h[1] = r1 * h1;
                size_t bon = (((size_t)(17 + kk) * NT + nt) * 32 + lane) * 16;
                bNn = *(const v16h*)(WnP + bon);
            }
            accN = WMMA_F16(a.v, bN, accN);
            if (kk < 15) { a = an; bN = bNn; }
        }
    }

    int mext = (lane >= 16) ? 8 : 0;
    int ncol = n0 + (lane & 15);
#pragma unroll
    for (int i = 0; i < 8; ++i) {
        int mrow = m0 + i + mext;
        size_t idx = (size_t)mrow * DD + ncol;
        _Float16 nv = (accN[i] >= 0.0f) ? (_Float16)1.0f : (_Float16)0.0f;
        _Float16 zv = zbuf[idx];
        _Float16 hn = (zv != (_Float16)0.0f) ? hprev[idx] : nv;
        hnext[idx] = hn;
        if (lens[mrow] - 1 == t) hg[idx] = hn;
    }
}

// ---------------------------------------------------------------------------
// Final dense on gathered hidden state: out = hg @ dense_W.T + b, f32 out.
// K = D = 512 (16 k-tiles), pipelined.
// ---------------------------------------------------------------------------
__global__ __launch_bounds__(128)
void dense_out_kernel(const _Float16* __restrict__ hg,
                      const _Float16* __restrict__ WdP,
                      const float* __restrict__ bias,
                      float* __restrict__ out) {
    int lane = threadIdx.x & 31;
    int wave = threadIdx.x >> 5;
    int m0 = blockIdx.x * 16;
    int n0 = blockIdx.y * 64 + wave * 16;
    int nt = n0 >> 4;
    int mrowA = m0 + (lane & 15);
    int k0 = (lane < 16) ? 0 : 8;
    const _Float16* arow = hg + (size_t)mrowA * DD + k0;

    v8f acc = {};
    {
        V16 a;
        a.h[0] = *(const v8h*)(arow);
        a.h[1] = *(const v8h*)(arow + 16);
        v16h b = *(const v16h*)(WdP + (((size_t)0 * NT + nt) * 32 + lane) * 16);
#pragma unroll
        for (int kk = 0; kk < 16; ++kk) {
            V16 an;
            v16h bn;
            if (kk < 15) {
                const _Float16* ar = arow + (kk + 1) * 32;
                an.h[0] = *(const v8h*)(ar);
                an.h[1] = *(const v8h*)(ar + 16);
                bn = *(const v16h*)(WdP + (((size_t)(kk + 1) * NT + nt) * 32 + lane) * 16);
            }
            acc = WMMA_F16(a.v, b, acc);
            if (kk < 15) { a = an; b = bn; }
        }
    }
    int mext = (lane >= 16) ? 8 : 0;
    int ncol = n0 + (lane & 15);
    float bv = bias[ncol];
#pragma unroll
    for (int i = 0; i < 8; ++i) {
        out[(size_t)(m0 + i + mext) * DD + ncol] = acc[i] + bv;
    }
}

// ---------------------------------------------------------------------------
// Host orchestration
// ---------------------------------------------------------------------------
extern "C" void kernel_launch(void* const* d_in, const int* in_sizes, int n_in,
                              void* d_out, int out_size, void* d_ws, size_t ws_size,
                              hipStream_t stream) {
    const int*   item_seq     = (const int*)d_in[0];
    const int*   item_seq_len = (const int*)d_in[1];
    const float* emb          = (const float*)d_in[2];
    const float* ln_gamma     = (const float*)d_in[3];
    const float* ln_beta      = (const float*)d_in[4];
    const float* Wr           = (const float*)d_in[5];
    const float* Wz           = (const float*)d_in[6];
    const float* Wn           = (const float*)d_in[7];
    const float* dense_W      = (const float*)d_in[8];
    const float* dense_b      = (const float*)d_in[9];
    float*       out          = (float*)d_out;

    char* ws = (char*)d_ws;
    size_t off = 0;
    auto take = [&](size_t bytes) -> char* {
        char* p = ws + off;
        off = (off + bytes + 255) & ~(size_t)255;
        return p;
    };
    _Float16* snn  = (_Float16*)take((size_t)LL * BB * DD * 2); // 52.4 MB
    _Float16* h0   = (_Float16*)take((size_t)BB * DD * 2);
    _Float16* h1   = (_Float16*)take((size_t)BB * DD * 2);
    _Float16* rbuf = (_Float16*)take((size_t)BB * DD * 2);
    _Float16* zbuf = (_Float16*)take((size_t)BB * DD * 2);
    _Float16* hg   = (_Float16*)take((size_t)BB * DD * 2);
    _Float16* WrP  = (_Float16*)take((size_t)2 * DD * DD * 2);  // 1 MB each
    _Float16* WzP  = (_Float16*)take((size_t)2 * DD * DD * 2);
    _Float16* WnP  = (_Float16*)take((size_t)2 * DD * DD * 2);
    _Float16* WdP  = (_Float16*)take((size_t)DD * DD * 2);

    // 1) pack weights into WMMA-fragment layout (f32 -> f16)
    {
        int totG = 2 * DD * DD;
        int totD = DD * DD;
        pack_w_kernel<<<(totG + 255) / 256, 256, 0, stream>>>(Wr, WrP, 2 * DD, DD);
        pack_w_kernel<<<(totG + 255) / 256, 256, 0, stream>>>(Wz, WzP, 2 * DD, DD);
        pack_w_kernel<<<(totG + 255) / 256, 256, 0, stream>>>(Wn, WnP, 2 * DD, DD);
        pack_w_kernel<<<(totD + 255) / 256, 256, 0, stream>>>(dense_W, WdP, DD, DD);
    }

    // 2) init h0 = 0, hg = 0
    {
        int n = BB * DD;
        zero_h16_kernel<<<(n + 255) / 256, 256, 0, stream>>>(h0, n);
        zero_h16_kernel<<<(n + 255) / 256, 256, 0, stream>>>(hg, n);
    }

    // 3) embedding + LayerNorm + LIF -> snn [l][b][d], f16
    embed_ln_lif_kernel<<<BB * LL, 256, 0, stream>>>(item_seq, emb, ln_gamma,
                                                     ln_beta, snn);

    // 4) spiking GRU recurrence, ping-pong h buffers
    dim3 g(BB / 16, DD / 64);
    for (int t = 0; t < LL; ++t) {
        _Float16* hp = (t & 1) ? h1 : h0;
        _Float16* hn = (t & 1) ? h0 : h1;
        gru_rz_kernel<<<g, 128, 0, stream>>>(snn, t, hp, WrP, WzP, rbuf, zbuf);
        gru_nh_kernel<<<g, 128, 0, stream>>>(snn, t, hp, rbuf, zbuf, WnP, hn, hg,
                                             item_seq_len);
    }

    // 5) dense on gathered hidden states -> [B, D] f32
    dense_out_kernel<<<g, 128, 0, stream>>>(hg, WdP, dense_b, out);
}
